// ActNet_21062519620386
// MI455X (gfx1250) — matrix operand order repaired and checked by reading, not compiled, not run on hardware
//
#include <hip/hip_runtime.h>
#include <math.h>

typedef __attribute__((ext_vector_type(16))) _Float16     v16h;
typedef __attribute__((ext_vector_type(8)))  float        v8f;
typedef __attribute__((ext_vector_type(4)))  unsigned int v4u;
typedef __attribute__((ext_vector_type(8)))  int          v8i;
typedef __attribute__((ext_vector_type(4)))  int          v4i;

#define BDIM 8192
#define NNODE 21
#define EE   256
#define MM   8
#define TT   (BDIM*NNODE)          /* 172032 tokens */
#define NSTEP (NNODE-1)            /* 20 decode steps */
#define INV_SCALE 0.17677669529663687f   /* 1/sqrt(32) */
#define NEGB (-1.0e9f)
#define BNEPS 1.0e-5f
#define CCLIP 10.0f

// ---------------------------------------------------------------------------
// wave32 reductions
// ---------------------------------------------------------------------------
__device__ __forceinline__ float wave_max(float v) {
#pragma unroll
  for (int off = 16; off > 0; off >>= 1) v = fmaxf(v, __shfl_xor(v, off, 32));
  return v;
}
__device__ __forceinline__ float wave_sum(float v) {
#pragma unroll
  for (int off = 16; off > 0; off >>= 1) v += __shfl_xor(v, off, 32);
  return v;
}
__device__ __forceinline__ void wave_argmax(float& v, int& i) {
#pragma unroll
  for (int off = 16; off > 0; off >>= 1) {
    float ov = __shfl_xor(v, off, 32);
    int   oi = __shfl_xor(i, off, 32);
    if (ov > v || (ov == v && oi < i)) { v = ov; i = oi; }
  }
}

// ---------------------------------------------------------------------------
// WMMA fragment loads (CDNA5 ISA 7.12.2 16-bit layouts, wave32)
// A: 16x32 (MxK). lanes 0-15: M=lane, elems 0-7 -> K=k0..k0+7, elems 8-15 ->
//    K=k0+16..k0+23. lanes 16-31: same M, K bases +8 / +24.
// ---------------------------------------------------------------------------
__device__ __forceinline__ v16h load_frag_a(const _Float16* __restrict__ A,
                                            int lda, int row0, int k0, int lane) {
  const int r  = row0 + (lane & 15);
  const int kb = k0 + ((lane >> 4) << 3);          // 0 or 8
  const _Float16* p = A + (size_t)r * lda + kb;
  v16h f;
#pragma unroll
  for (int i = 0; i < 8; ++i) f[i] = p[i];
#pragma unroll
  for (int i = 0; i < 8; ++i) f[8 + i] = p[16 + i];
  return f;
}
// B: 32x16 (KxN) from transposed weights (N x K row-major, global or LDS).
// lane column = c0+(lane&15); lanes 0-15 hold K=k0..k0+7 (elems 0-7) and
// K=k0+16..k0+23 (elems 8-15); lanes 16-31 same with K bases +8 / +24.
__device__ __forceinline__ v16h load_frag_b(const _Float16* Wt, int ldb, int c0,
                                            int k0, int lane) {
  const int c  = c0 + (lane & 15);
  const int kb = k0 + ((lane >> 4) << 3);          // 0 or 8
  const _Float16* p = Wt + (size_t)c * ldb + kb;
  v16h f;
#pragma unroll
  for (int i = 0; i < 8; ++i) f[i] = p[i];
#pragma unroll
  for (int i = 0; i < 8; ++i) f[8 + i] = p[16 + i];
  return f;
}

// ---------------------------------------------------------------------------
// GEMM: out(T x Nc) = A(T x K f16) * Wt(Nc x K f16)^T + bias (+relu)(+resid)
// 128 threads = 4 waves; wave computes 16 rows x (16*CT) cols.
// TDM=true: block's weight tile (16*CT rows of Wt x K) is DMAd into LDS by the
// Tensor Data Mover (2D D#, data_size=2B), then B fragments come from LDS.
// Requires T%64==0, Nc%(16*CT)==0, K%32==0.
// ---------------------------------------------------------------------------
template<int CT, bool TDM, bool RELU, bool RESID, bool OUT32, bool OUT16>
__global__ __launch_bounds__(128)
void gemm_f16_wmma(const _Float16* __restrict__ A,
                   const _Float16* __restrict__ Wt,
                   const float* __restrict__ bias,
                   const float* resid,      // may alias out32
                   float* out32,
                   _Float16* __restrict__ out16,
                   int T, int K, int Nc) {
  const int lane = threadIdx.x & 31;
  const int wave = threadIdx.x >> 5;
  const int row0 = blockIdx.x * 64 + wave * 16;
  const int colB = blockIdx.y * (16 * CT);
  extern __shared__ _Float16 bt[];                 // 16*CT x K f16 (TDM only)

  if (TDM) {
    if (wave == 0) {
      const unsigned long long ga =
          (unsigned long long)(uintptr_t)(Wt + (size_t)colB * K);
      const unsigned int ldsb = (unsigned int)(uintptr_t)(void*)bt;
      const unsigned int rows = 16 * CT;
      v4u g0; v8i g1; v4i g2 = {0, 0, 0, 0}; v4i g3 = {0, 0, 0, 0};
      v8i g4 = {0, 0, 0, 0, 0, 0, 0, 0};
      // D# group0: count=1 | lds_addr | global_addr[56:0] | type=2
      g0[0] = 1u;
      g0[1] = ldsb;
      g0[2] = (unsigned int)ga;
      g0[3] = (unsigned int)((ga >> 32) & 0x1FFFFFFu) | (2u << 30);
      // D# group1: data_size=1(2B); tensor_dim0=K, tensor_dim1=rows;
      // tile_dim0=K, tile_dim1=rows; tensor_dim0_stride=K
      g1[0] = (int)(1u << 16);
      g1[1] = (int)((unsigned)(K & 0xFFFF) << 16);
      g1[2] = (int)(((unsigned)K >> 16) | ((rows & 0xFFFFu) << 16));
      g1[3] = (int)((unsigned)(K & 0xFFFF) << 16);
      g1[4] = (int)rows;
      g1[5] = K;
      g1[6] = 0; g1[7] = 0;
      __builtin_amdgcn_tensor_load_to_lds(g0, g1, g2, g3, g4, 0);
      __builtin_amdgcn_s_wait_tensorcnt(0);
    }
    __syncthreads();
  }

  v8f acc[CT];
#pragma unroll
  for (int ct = 0; ct < CT; ++ct) acc[ct] = v8f{};
  for (int k0 = 0; k0 < K; k0 += 32) {
    if (!TDM && k0 + 32 < K)
      __builtin_prefetch(A + (size_t)(row0 + (lane & 15)) * K + k0 + 32, 0, 1);
    v16h a = load_frag_a(A, K, row0, k0, lane);
#pragma unroll
    for (int ct = 0; ct < CT; ++ct) {
      v16h b = TDM ? load_frag_b(bt, K, ct * 16, k0, lane)
                   : load_frag_b(Wt, K, colB + ct * 16, k0, lane);
      acc[ct] = __builtin_amdgcn_wmma_f32_16x16x32_f16(
          false, a, false, b, (short)0, acc[ct], false, false);
    }
  }
  // C/D layout: VGPR r -> row (lane<16 ? r : 8+r), col = lane&15
  const int nn = lane & 15;
  const int rb = row0 + ((lane >> 4) << 3);
#pragma unroll
  for (int ct = 0; ct < CT; ++ct) {
    const int n   = colB + ct * 16 + nn;
    const float bia = bias[n];
#pragma unroll
    for (int r = 0; r < 8; ++r) {
      const size_t o = (size_t)(rb + r) * Nc + n;
      float v = acc[ct][r] + bia;
      if (RELU)  v = fmaxf(v, 0.0f);
      if (RESID) v += resid[o];
      if (OUT32) out32[o] = v;
      if (OUT16) out16[o] = (_Float16)v;
    }
  }
}

// ---------------------------------------------------------------------------
// elementwise / small kernels
// ---------------------------------------------------------------------------
__global__ void pack_wt_kernel(const float* __restrict__ W, _Float16* __restrict__ Wt,
                               int din, int dout, int total) {
  int i = blockIdx.x * blockDim.x + threadIdx.x;
  if (i >= total) return;
  int r = i / din, c = i - r * din;                // Wt[r][c] = W[c][r]
  Wt[i] = (_Float16)W[(size_t)c * dout + r];
}

__global__ void concat3_kernel(const float* a, const float* b, const float* c,
                               float* o) {
  int i = blockIdx.x * blockDim.x + threadIdx.x;
  if (i >= 3 * EE) return;
  o[i] = (i < EE) ? a[i] : (i < 2 * EE) ? b[i - EE] : c[i - 2 * EE];
}

__global__ void embed_kernel(const float* __restrict__ x_, const float* __restrict__ W,
                             const float* __restrict__ bvec, float* __restrict__ X,
                             _Float16* __restrict__ Xh, int total) {
  int i = blockIdx.x * blockDim.x + threadIdx.x;
  if (i >= total) return;
  int t = i >> 8, e = i & 255;
  const float* xr = x_ + (size_t)t * 3;
  float v = bvec[e] + xr[0] * W[e] + xr[1] * W[EE + e] + xr[2] * W[2 * EE + e];
  X[i] = v; Xh[i] = (_Float16)v;
}

// BN stats: deterministic two-stage (chunk of 256 rows per block, channel/thread)
__global__ __launch_bounds__(256)
void bn_partial_kernel(const float* __restrict__ X, float* __restrict__ part) {
  int e = threadIdx.x, c = blockIdx.x;
  size_t base = (size_t)c * 256 * EE + e;
  float s = 0.f, s2 = 0.f;
  for (int r = 0; r < 256; ++r) {
    float v = X[base + (size_t)r * EE];
    s += v; s2 += v * v;
  }
  part[(size_t)c * 512 + e] = s;
  part[(size_t)c * 512 + 256 + e] = s2;
}
__global__ __launch_bounds__(256)
void bn_finalize_kernel(const float* __restrict__ part, const float* __restrict__ g,
                        const float* __restrict__ bb, float* __restrict__ ss,
                        int chunks, float invT) {
  int e = threadIdx.x;
  float s = 0.f, s2 = 0.f;
  for (int c = 0; c < chunks; ++c) {
    s  += part[(size_t)c * 512 + e];
    s2 += part[(size_t)c * 512 + 256 + e];
  }
  float mean = s * invT;
  float var  = s2 * invT - mean * mean;            // biased var (jnp.var)
  float sc   = g[e] * rsqrtf(var + BNEPS);
  ss[e] = sc; ss[EE + e] = bb[e] - mean * sc;
}
__global__ void bn_apply_kernel(float* __restrict__ X, _Float16* __restrict__ Xh,
                                const float* __restrict__ ss, int total) {
  int i = blockIdx.x * blockDim.x + threadIdx.x;
  if (i >= total) return;
  int c = i & 255;
  float v = X[i] * ss[c] + ss[EE + c];
  X[i] = v; Xh[i] = (_Float16)v;
}

__global__ void ave_kernel(const float* __restrict__ X, float* __restrict__ ave,
                           int total) {
  int i = blockIdx.x * blockDim.x + threadIdx.x;
  if (i >= total) return;
  int b = i >> 8, e = i & 255;
  float s = 0.f;
  for (int n = 0; n < NNODE; ++n) s += X[((size_t)b * NNODE + n) * EE + e];
  ave[i] = s * (1.0f / (float)NNODE);
}

// ---------------------------------------------------------------------------
// encoder attention: one wave per (b, m); 21x21 softmax in f32
// ---------------------------------------------------------------------------
__global__ __launch_bounds__(32)
void enc_attn_kernel(const _Float16* __restrict__ qkv, _Float16* __restrict__ zout) {
  const int b = blockIdx.x >> 3, m = blockIdx.x & 7;
  const int lane = threadIdx.x;
  __shared__ _Float16 qs[NNODE][32], ks[NNODE][32], vs[NNODE][32];
  __shared__ float arow[NNODE];
  const size_t base = (size_t)b * NNODE * 768 + m * 32;
  for (int n = 0; n < NNODE; ++n) {
    size_t ro = base + (size_t)n * 768;
    qs[n][lane] = qkv[ro + lane];
    ks[n][lane] = qkv[ro + 256 + lane];
    vs[n][lane] = qkv[ro + 512 + lane];
  }
  __syncthreads();
  for (int i = 0; i < NNODE; ++i) {
    float s = -3.0e38f;
    if (lane < NNODE) {
      float acc = 0.f;
#pragma unroll
      for (int d = 0; d < 32; ++d) acc += (float)qs[i][d] * (float)ks[lane][d];
      s = acc * INV_SCALE;
    }
    float mx = wave_max(s);
    float e  = (lane < NNODE) ? __expf(s - mx) : 0.f;
    float sm = wave_sum(e);
    if (lane < NNODE) arow[lane] = e / sm;
    __syncthreads();
    float z = 0.f;
    for (int j = 0; j < NNODE; ++j) z += arow[j] * (float)vs[j][lane];
    zout[((size_t)b * NNODE + i) * EE + m * 32 + lane] = (_Float16)z;
    __syncthreads();
  }
}

// ---------------------------------------------------------------------------
// decode
// ---------------------------------------------------------------------------
__global__ void init_decode_kernel(int* __restrict__ mask, int* __restrict__ idx,
                                   float* __restrict__ dist, int total) {
  int i = blockIdx.x * blockDim.x + threadIdx.x;
  if (i >= total) return;
  mask[i] = 0;
  if (i < BDIM) { idx[i] = 0; dist[i] = 0.f; }
}

__global__ void gi_kernel(const float* __restrict__ ave, const float* __restrict__ X,
                          int* __restrict__ mask, const int* __restrict__ idx,
                          _Float16* __restrict__ gih, int total) {
  int i = blockIdx.x * blockDim.x + threadIdx.x;
  if (i >= total) return;
  int b = i >> 9, e = i & 511;
  int cur = idx[b];
  if (e == 0) mask[b * NNODE + cur] = 1;           // mask |= (arange == idx)
  float v = (e < EE) ? ave[(size_t)b * EE + e]
                     : X[((size_t)b * NNODE + cur) * EE + (e - EE)];
  gih[i] = (_Float16)v;
}

__global__ __launch_bounds__(32)
void dec_attn_kernel(const _Float16* __restrict__ q4, const _Float16* __restrict__ k4,
                     const _Float16* __restrict__ v4, const int* __restrict__ mask,
                     _Float16* __restrict__ zout) {
  const int b = blockIdx.x >> 3, m = blockIdx.x & 7;
  const int lane = threadIdx.x;
  __shared__ _Float16 ks[NNODE][32], vs[NNODE][32];
  __shared__ float qf[32], arow[NNODE];
  qf[lane] = (float)q4[(size_t)b * EE + m * 32 + lane];
  for (int n = 0; n < NNODE; ++n) {
    size_t o = ((size_t)b * NNODE + n) * EE + m * 32 + lane;
    ks[n][lane] = k4[o]; vs[n][lane] = v4[o];
  }
  __syncthreads();
  float s = -3.0e38f;
  if (lane < NNODE) {
    float acc = 0.f;
#pragma unroll
    for (int d = 0; d < 32; ++d) acc += qf[d] * (float)ks[lane][d];
    s = acc * INV_SCALE;
    if (mask[b * NNODE + lane]) s = NEGB;
  }
  float mx = wave_max(s);
  float e  = (lane < NNODE) ? __expf(s - mx) : 0.f;
  float sm = wave_sum(e);
  if (lane < NNODE) arow[lane] = e / sm;
  __syncthreads();
  float z = 0.f;
  for (int j = 0; j < NNODE; ++j) z += arow[j] * (float)vs[j][lane];
  zout[(size_t)b * EE + m * 32 + lane] = (_Float16)z;
}

__global__ __launch_bounds__(32)
void pointer_kernel(const _Float16* __restrict__ q5, const _Float16* __restrict__ k5,
                    const int* __restrict__ mask, const float* __restrict__ x_,
                    int* __restrict__ idx, float* __restrict__ seq,
                    float* __restrict__ pro, float* __restrict__ dist, int step) {
  const int b = blockIdx.x;
  const int lane = threadIdx.x;
  __shared__ float q5s[EE];
  for (int i = lane; i < EE; i += 32) q5s[i] = (float)q5[(size_t)b * EE + i];
  __syncthreads();
  float logit = -3.0e38f;
  if (lane < NNODE) {
    const _Float16* kp = k5 + ((size_t)b * NNODE + lane) * EE;
    float acc = 0.f;
    for (int d = 0; d < EE; ++d) acc += q5s[d] * (float)kp[d];
    logit = tanhf(acc * INV_SCALE) * CCLIP;
    if (mask[b * NNODE + lane]) logit = NEGB;
  }
  float mx = wave_max(logit);
  float e  = (lane < NNODE) ? __expf(logit - mx) : 0.f;
  float sm = wave_sum(e);
  float p  = (lane < NNODE) ? (e / sm) : -1.f;
  int   bi = (lane < NNODE) ? lane : 1000;
  wave_argmax(p, bi);                               // first-max tie-break
  if (lane == 0) {
    int cur = idx[b];
    float d0 = 0.f;
    if (bi >= 3) {                                  // dis[:, :, j<A] == 0
      float dx = x_[((size_t)b * NNODE + bi) * 3 + 0] - x_[((size_t)b * NNODE + cur) * 3 + 0];
      float dy = x_[((size_t)b * NNODE + bi) * 3 + 1] - x_[((size_t)b * NNODE + cur) * 3 + 1];
      d0 = sqrtf(dx * dx + dy * dy);
    }
    dist[b] += d0;
    seq[(size_t)b * NSTEP + step] = (float)bi;
    pro[(size_t)b * NSTEP + step] = p;
    idx[b] = bi;
  }
}

// ---------------------------------------------------------------------------
// host launch
// ---------------------------------------------------------------------------
struct Params {
  const float *embW, *embB;
  const float *wqW[3], *wqB[3], *wkW[3], *wkB[3], *wvW[3], *wvB[3], *wW[3], *wB[3];
  const float *f1W[3], *f1B[3], *f2W[3], *f2B[3];
  const float *bn1g[3], *bn1b[3], *bn2g[3], *bn2b[3];
  const float *wq4W, *wq4B, *wk4W, *wk4B, *wv4W, *wv4B;
  const float *w4W, *w4B, *wq5W, *wq5B, *wk5W, *wk5B;
  const float *x_;
};

extern "C" void kernel_launch(void* const* d_in, const int* in_sizes, int n_in,
                              void* d_out, int out_size, void* d_ws, size_t ws_size,
                              hipStream_t stream) {
  auto F = [&](int i) { return (const float*)d_in[i]; };
  Params P;
  if (n_in >= 64 && in_sizes[0] == 1) {
    // JAX sorted-key flatten order: is_train, params(sorted), x_
    for (int l = 0; l < 3; ++l) {
      P.bn1g[l] = F(1 + 4 * l); P.bn1b[l] = F(2 + 4 * l);
      P.bn2g[l] = F(3 + 4 * l); P.bn2b[l] = F(4 + 4 * l);
    }
    P.embW = F(13); P.embB = F(14);
    for (int l = 0; l < 3; ++l) {
      P.f1W[l] = F(15 + 4 * l); P.f1B[l] = F(16 + 4 * l);
      P.f2W[l] = F(17 + 4 * l); P.f2B[l] = F(18 + 4 * l);
    }
    for (int l = 0; l < 3; ++l) { P.wW[l] = F(27 + 2 * l); P.wB[l] = F(28 + 2 * l); }
    P.w4W = F(33); P.w4B = F(34);
    for (int l = 0; l < 3; ++l) { P.wkW[l] = F(35 + 2 * l); P.wkB[l] = F(36 + 2 * l); }
    P.wk4W = F(41); P.wk4B = F(42); P.wk5W = F(43); P.wk5B = F(44);
    for (int l = 0; l < 3; ++l) { P.wqW[l] = F(45 + 2 * l); P.wqB[l] = F(46 + 2 * l); }
    P.wq4W = F(51); P.wq4B = F(52); P.wq5W = F(53); P.wq5B = F(54);
    for (int l = 0; l < 3; ++l) { P.wvW[l] = F(55 + 2 * l); P.wvB[l] = F(56 + 2 * l); }
    P.wv4W = F(61); P.wv4B = F(62);
    P.x_ = F(63);
  } else {
    // insertion order: x_, embedding, per-layer {wq,wk,wv,w,ffc1,ffc2,bn1,bn2}, wq4..wk5
    P.x_ = F(0); P.embW = F(1); P.embB = F(2);
    for (int l = 0; l < 3; ++l) {
      int base = 3 + 16 * l;
      P.wqW[l] = F(base + 0);  P.wqB[l] = F(base + 1);
      P.wkW[l] = F(base + 2);  P.wkB[l] = F(base + 3);
      P.wvW[l] = F(base + 4);  P.wvB[l] = F(base + 5);
      P.wW[l]  = F(base + 6);  P.wB[l]  = F(base + 7);
      P.f1W[l] = F(base + 8);  P.f1B[l] = F(base + 9);
      P.f2W[l] = F(base + 10); P.f2B[l] = F(base + 11);
      P.bn1g[l] = F(base + 12); P.bn1b[l] = F(base + 13);
      P.bn2g[l] = F(base + 14); P.bn2b[l] = F(base + 15);
    }
    P.wq4W = F(51); P.wq4B = F(52); P.wk4W = F(53); P.wk4B = F(54);
    P.wv4W = F(55); P.wv4B = F(56); P.w4W = F(57);  P.w4B = F(58);
    P.wq5W = F(59); P.wq5B = F(60); P.wk5W = F(61); P.wk5B = F(62);
  }

  // ---- workspace layout (bump allocator, 256B aligned) ----
  char* wp = (char*)d_ws;
  auto alloc = [&](size_t bytes) -> void* {
    void* p = wp; wp += (bytes + 255) & ~(size_t)255; return p;
  };
  const int CH = TT / 256;  // 672 BN chunks
  float*     X    = (float*)alloc((size_t)TT * EE * 4);
  _Float16*  Xh   = (_Float16*)alloc((size_t)TT * EE * 2);
  _Float16*  QKV  = (_Float16*)alloc((size_t)TT * 768 * 2);   // reused as K4|V4|K5
  _Float16*  SCR  = (_Float16*)alloc((size_t)TT * EE * 2);    // Zh / FF hidden
  float*     PART = (float*)alloc((size_t)CH * 512 * 4);
  float*     SS   = (float*)alloc(2 * EE * 4);
  float*     AVE  = (float*)alloc((size_t)BDIM * EE * 4);
  _Float16*  GIH  = (_Float16*)alloc((size_t)BDIM * 512 * 2);
  _Float16*  Q4H  = (_Float16*)alloc((size_t)BDIM * EE * 2);
  _Float16*  ZH   = (_Float16*)alloc((size_t)BDIM * EE * 2);
  _Float16*  Z2H  = (_Float16*)alloc((size_t)BDIM * EE * 2);
  _Float16*  Q5H  = (_Float16*)alloc((size_t)BDIM * EE * 2);
  int*       MASK = (int*)alloc((size_t)BDIM * NNODE * 4);
  int*       IDX  = (int*)alloc((size_t)BDIM * 4);
  _Float16*  WQKVt[3]; float* BQKV[3];
  _Float16 *WWt[3], *WF1t[3], *WF2t[3];
  for (int l = 0; l < 3; ++l) {
    WQKVt[l] = (_Float16*)alloc((size_t)768 * EE * 2);
    BQKV[l]  = (float*)alloc(768 * 4);
    WWt[l]   = (_Float16*)alloc((size_t)EE * EE * 2);
    WF1t[l]  = (_Float16*)alloc((size_t)EE * EE * 2);
    WF2t[l]  = (_Float16*)alloc((size_t)EE * EE * 2);
  }
  _Float16* WK4t = (_Float16*)alloc((size_t)EE * EE * 2);
  _Float16* WV4t = (_Float16*)alloc((size_t)EE * EE * 2);
  _Float16* WK5t = (_Float16*)alloc((size_t)EE * EE * 2);
  _Float16* WQ4t = (_Float16*)alloc((size_t)EE * 512 * 2);
  _Float16* WW4t = (_Float16*)alloc((size_t)EE * EE * 2);
  _Float16* WQ5t = (_Float16*)alloc((size_t)EE * EE * 2);
  if ((size_t)(wp - (char*)d_ws) > ws_size) return;  // insufficient scratch

  float* seq  = (float*)d_out;
  float* pro  = seq + (size_t)BDIM * NSTEP;
  float* dist = pro + (size_t)BDIM * NSTEP;

  auto pack = [&](const float* W, _Float16* Wt, int din, int dout) {
    int n = din * dout;
    pack_wt_kernel<<<(n + 255) / 256, 256, 0, stream>>>(W, Wt, din, dout, n);
  };

  // ---- pack all weights (transposed f16) ----
  for (int l = 0; l < 3; ++l) {
    pack(P.wqW[l], WQKVt[l] + 0 * EE * EE, EE, EE);
    pack(P.wkW[l], WQKVt[l] + 1 * EE * EE, EE, EE);
    pack(P.wvW[l], WQKVt[l] + 2 * EE * EE, EE, EE);
    concat3_kernel<<<3, 256, 0, stream>>>(P.wqB[l], P.wkB[l], P.wvB[l], BQKV[l]);
    pack(P.wW[l], WWt[l], EE, EE);
    pack(P.f1W[l], WF1t[l], EE, EE);
    pack(P.f2W[l], WF2t[l], EE, EE);
  }
  pack(P.wk4W, WK4t, EE, EE); pack(P.wv4W, WV4t, EE, EE);
  pack(P.wk5W, WK5t, EE, EE); pack(P.wq4W, WQ4t, 512, EE);
  pack(P.w4W,  WW4t, EE, EE); pack(P.wq5W, WQ5t, EE, EE);

  // ---- embedding ----
  {
    int n = TT * EE;
    embed_kernel<<<(n + 255) / 256, 256, 0, stream>>>(P.x_, P.embW, P.embB, X, Xh, n);
  }

  const int elemN = TT * EE;
  const dim3 gBlk(128);
  const size_t SHB = (size_t)128 * EE * 2;  // 64KB LDS weight tile (K=256)
  auto bn = [&](const float* g, const float* b) {
    bn_partial_kernel<<<CH, 256, 0, stream>>>(X, PART);
    bn_finalize_kernel<<<1, 256, 0, stream>>>(PART, g, b, SS, CH, 1.0f / (float)TT);
    bn_apply_kernel<<<(elemN + 255) / 256, 256, 0, stream>>>(X, Xh, SS, elemN);
  };

  // ---- 3 encoder layers (TDM-staged weight tiles, 16x128 per wave) ----
  for (int l = 0; l < 3; ++l) {
    gemm_f16_wmma<8, true, false, false, false, true>
        <<<dim3(TT / 64, 768 / 128), gBlk, SHB, stream>>>(
        Xh, WQKVt[l], BQKV[l], nullptr, nullptr, QKV, TT, EE, 768);
    enc_attn_kernel<<<BDIM * MM, 32, 0, stream>>>(QKV, SCR);
    gemm_f16_wmma<8, true, false, true, true, false>
        <<<dim3(TT / 64, EE / 128), gBlk, SHB, stream>>>(
        SCR, WWt[l], P.wB[l], X, X, nullptr, TT, EE, EE);
    bn(P.bn1g[l], P.bn1b[l]);
    gemm_f16_wmma<8, true, true, false, false, true>
        <<<dim3(TT / 64, EE / 128), gBlk, SHB, stream>>>(
        Xh, WF1t[l], P.f1B[l], nullptr, nullptr, SCR, TT, EE, EE);
    gemm_f16_wmma<8, true, false, true, true, false>
        <<<dim3(TT / 64, EE / 128), gBlk, SHB, stream>>>(
        SCR, WF2t[l], P.f2B[l], X, X, nullptr, TT, EE, EE);
    bn(P.bn2g[l], P.bn2b[l]);
  }

  // ---- decode prep: ave, k4, v4, k5 (QKV region reused) ----
  _Float16* K4 = QKV;
  _Float16* V4 = QKV + (size_t)TT * EE;
  _Float16* K5 = QKV + (size_t)2 * TT * EE;
  {
    int n = BDIM * EE;
    ave_kernel<<<(n + 255) / 256, 256, 0, stream>>>(X, AVE, n);
  }
  gemm_f16_wmma<8, true, false, false, false, true>
      <<<dim3(TT / 64, EE / 128), gBlk, SHB, stream>>>(
      Xh, WK4t, P.wk4B, nullptr, nullptr, K4, TT, EE, EE);
  gemm_f16_wmma<8, true, false, false, false, true>
      <<<dim3(TT / 64, EE / 128), gBlk, SHB, stream>>>(
      Xh, WV4t, P.wv4B, nullptr, nullptr, V4, TT, EE, EE);
  gemm_f16_wmma<8, true, false, false, false, true>
      <<<dim3(TT / 64, EE / 128), gBlk, SHB, stream>>>(
      Xh, WK5t, P.wk5B, nullptr, nullptr, K5, TT, EE, EE);

  {
    int n = BDIM * NNODE;
    init_decode_kernel<<<(n + 255) / 256, 256, 0, stream>>>(MASK, IDX, dist, n);
  }

  // ---- 20 greedy decode steps (small GEMMs: direct-global B, CT=2) ----
  for (int step = 0; step < NSTEP; ++step) {
    int n = BDIM * 512;
    gi_kernel<<<(n + 255) / 256, 256, 0, stream>>>(AVE, X, MASK, IDX, GIH, n);
    gemm_f16_wmma<2, false, false, false, false, true>
        <<<dim3(BDIM / 64, EE / 32), gBlk, 0, stream>>>(
        GIH, WQ4t, P.wq4B, nullptr, nullptr, Q4H, BDIM, 512, EE);
    dec_attn_kernel<<<BDIM * MM, 32, 0, stream>>>(Q4H, K4, V4, MASK, ZH);
    gemm_f16_wmma<2, false, false, false, false, true>
        <<<dim3(BDIM / 64, EE / 32), gBlk, 0, stream>>>(
        ZH, WW4t, P.w4B, nullptr, nullptr, Z2H, BDIM, EE, EE);
    gemm_f16_wmma<2, false, false, false, false, true>
        <<<dim3(BDIM / 64, EE / 32), gBlk, 0, stream>>>(
        Z2H, WQ5t, P.wq5B, nullptr, nullptr, Q5H, BDIM, EE, EE);
    pointer_kernel<<<BDIM, 32, 0, stream>>>(Q5H, K5, MASK, P.x_, IDX, seq, pro, dist, step);
  }
}
